// EnhancedGCNEncoder_86912958202426
// MI455X (gfx1250) — compile-verified, matrix operand order checked by reading, not run on hardware
//
#include <hip/hip_runtime.h>
#include <hip/hip_bf16.h>

typedef __attribute__((ext_vector_type(16))) __bf16 v16bf;
typedef __attribute__((ext_vector_type(8)))  float  v8f;

#define H128 128
#define HEADS 4
#define DHD 32
#define EPSBN 1e-5f
#define NEG_SLOPE 0.2f

// ---------------------------------------------------------------------------
// Weight packing: W[M,128] f32 row-major -> bf16 in exact WMMA B-fragment
// order: out[((ktile*8 + wtile)*32 + lane)*16 + j], j = 2v+p,
// k = ktile*32 + (v<4 ? kbase+2v : 16+kbase+2(v-4)) + p, kbase = (lane>>4)*8,
// col = wtile*16 + (lane&15).
// ---------------------------------------------------------------------------
__global__ void pack_w_bf16(const float* __restrict__ W, __bf16* __restrict__ out, int total) {
    int id = blockIdx.x * blockDim.x + threadIdx.x;
    if (id >= total) return;
    int j     = id & 15;
    int lane  = (id >> 4) & 31;
    int wt    = (id >> 9) & 7;
    int ktile = id >> 12;
    int v = j >> 1, p = j & 1;
    int kbase = (lane >> 4) << 3;
    int k0 = (v < 4) ? (kbase + 2 * v) : (16 + kbase + 2 * (v - 4));
    int k   = ktile * 32 + k0 + p;
    int col = (wt << 4) + (lane & 15);
    out[id] = (__bf16)W[(size_t)k * H128 + col];
}

// ---------------------------------------------------------------------------
// WMMA GEMM: C[rows,128] = A[rows,lda] @ W[M,128] (+bias) (optional relu)
// A: f32 row-major, converted to bf16 on the fly (vectorized b128 loads)
// Wp: pre-packed bf16 B fragments (one v16bf load per wave per k-step)
// block = 256 threads = 8 waves; wave w -> column tile [16w,16w+16)
// grid.x = rows/16 (rows always multiple of 16 here)
// ---------------------------------------------------------------------------
template <int HAS_BIAS, int RELU>
__global__ __launch_bounds__(256)
void wmma_gemm128(const float* __restrict__ A, int lda,
                  const __bf16* __restrict__ Wp,
                  const float* __restrict__ bias,
                  float* __restrict__ C,
                  int M)
{
    const int wave    = threadIdx.x >> 5;
    const int lane    = threadIdx.x & 31;
    const int rowbase = blockIdx.x << 4;
    const int colbase = wave << 4;
    const int rlocal  = lane & 15;
    const int kbase   = (lane >> 4) << 3;   // 0 or 8

    const float*  arow = A + (size_t)(rowbase + rlocal) * lda;
    const __bf16* bptr = Wp + ((size_t)wave * 32 + lane) * 16;

    v8f acc = {};
    for (int kk = 0; kk < M; kk += 32) {
        const float4 a0 = *(const float4*)(arow + kk + kbase);
        const float4 a1 = *(const float4*)(arow + kk + kbase + 4);
        const float4 a2 = *(const float4*)(arow + kk + 16 + kbase);
        const float4 a3 = *(const float4*)(arow + kk + 16 + kbase + 4);
        const v16bf  bf = *(const v16bf*)bptr;
        bptr += 8 * 32 * 16;   // advance one k-tile

        v16bf af;
        af[0]  = (__bf16)a0.x; af[1]  = (__bf16)a0.y;
        af[2]  = (__bf16)a0.z; af[3]  = (__bf16)a0.w;
        af[4]  = (__bf16)a1.x; af[5]  = (__bf16)a1.y;
        af[6]  = (__bf16)a1.z; af[7]  = (__bf16)a1.w;
        af[8]  = (__bf16)a2.x; af[9]  = (__bf16)a2.y;
        af[10] = (__bf16)a2.z; af[11] = (__bf16)a2.w;
        af[12] = (__bf16)a3.x; af[13] = (__bf16)a3.y;
        af[14] = (__bf16)a3.z; af[15] = (__bf16)a3.w;

        acc = __builtin_amdgcn_wmma_f32_16x16x32_bf16(
                  false, af, false, bf, (short)0, acc, false, false);
    }

    const int cl     = lane & 15;
    const int rowadd = (lane < 16) ? 0 : 8;
    const float bv   = HAS_BIAS ? bias[colbase + cl] : 0.0f;
#pragma unroll
    for (int v = 0; v < 8; ++v) {
        float o = acc[v] + bv;
        if (RELU) o = fmaxf(o, 0.0f);
        C[(size_t)(rowbase + rowadd + v) * H128 + colbase + cl] = o;
    }
}

// ---------------------------------------------------------------------------
// elementwise helpers
// ---------------------------------------------------------------------------
__global__ void fill_f32(float* __restrict__ p, float v, int n) {
    int i = blockIdx.x * blockDim.x + threadIdx.x;
    if (i < n) p[i] = v;
}

__global__ void deg_edge(const int* __restrict__ dst, float* __restrict__ deg, int E) {
    int e = blockIdx.x * blockDim.x + threadIdx.x;
    if (e < E) atomicAdd(&deg[dst[e]], 1.0f);
}

__global__ void deg_to_dinv(float* __restrict__ d, int n) {
    int i = blockIdx.x * blockDim.x + threadIdx.x;
    if (i < n) { float v = d[i]; d[i] = (v > 0.0f) ? rsqrtf(v) : 0.0f; }
}

// agg[i,f] = hW[i,f] * dinv[i]^2   (self-loop term)
__global__ void gcn_init(const float* __restrict__ hW, const float* __restrict__ dinv,
                         float* __restrict__ agg, int total) {
    int id = blockIdx.x * blockDim.x + threadIdx.x;
    if (id >= total) return;
    int n = id >> 7;
    float di = dinv[n];
    agg[id] = hW[id] * di * di;
}

// agg[d,f] += hW[s,f] * dinv[s]*dinv[d]
__global__ void gcn_edge(const int* __restrict__ src, const int* __restrict__ dst,
                         const float* __restrict__ hW, const float* __restrict__ dinv,
                         float* __restrict__ agg, int total) {
    int id = blockIdx.x * blockDim.x + threadIdx.x;
    if (id >= total) return;
    int e = id >> 7, f = id & 127;
    int s = src[e], d = dst[e];
    atomicAdd(&agg[(size_t)d * H128 + f], hW[(size_t)s * H128 + f] * dinv[s] * dinv[d]);
}

// out = relu(bn(agg + bias)) + identity
__global__ void finalize_bn(const float* __restrict__ agg, const float* __restrict__ bias,
                            const float* __restrict__ g, const float* __restrict__ bb,
                            const float* __restrict__ rm, const float* __restrict__ rv,
                            const float* __restrict__ ident, float* __restrict__ out,
                            int total) {
    int id = blockIdx.x * blockDim.x + threadIdx.x;
    if (id >= total) return;
    int f = id & 127;
    float x = agg[id] + bias[f];
    x = (x - rm[f]) * rsqrtf(rv[f] + EPSBN) * g[f] + bb[f];
    x = fmaxf(x, 0.0f);
    out[id] = x + ident[id];
}

// ---------------------------------------------------------------------------
// GAT pieces
// ---------------------------------------------------------------------------
__global__ void gat_alpha(const float* __restrict__ hW,
                          const float* __restrict__ a_src, const float* __restrict__ a_dst,
                          float* __restrict__ as_, float* __restrict__ ad_, int total) {
    int id = blockIdx.x * blockDim.x + threadIdx.x;
    if (id >= total) return;
    int n = id >> 2, h = id & 3;
    const float* row = hW + (size_t)n * H128 + h * DHD;
    float s1 = 0.0f, s2 = 0.0f;
#pragma unroll 8
    for (int j = 0; j < DHD; ++j) {
        float v = row[j];
        s1 += v * a_src[h * DHD + j];
        s2 += v * a_dst[h * DHD + j];
    }
    as_[id] = s1; ad_[id] = s2;
}

__device__ __forceinline__ unsigned enc_f(float f) {
    unsigned b = __float_as_uint(f);
    return (b & 0x80000000u) ? ~b : (b | 0x80000000u);
}
__device__ __forceinline__ float dec_f(unsigned u) {
    unsigned b = (u & 0x80000000u) ? (u & 0x7FFFFFFFu) : ~u;
    return __uint_as_float(b);
}
__device__ __forceinline__ float lrelu(float v) { return v > 0.0f ? v : NEG_SLOPE * v; }

// segment max over edges+selfloops, per head (monotone uint encoding)
__global__ void gat_max(const int* __restrict__ src, const int* __restrict__ dst,
                        const float* __restrict__ as_, const float* __restrict__ ad_,
                        unsigned* __restrict__ menc, int E, int total) {
    int id = blockIdx.x * blockDim.x + threadIdx.x;
    if (id >= total) return;
    int e = id >> 2, h = id & 3;
    int s, d;
    if (e < E) { s = src[e]; d = dst[e]; } else { s = d = e - E; }
    float v = lrelu(as_[s * 4 + h] + ad_[d * 4 + h]);
    atomicMax(&menc[d * 4 + h], enc_f(v));
}

// ex = exp(e - m[d]); denom[d] += ex; stash ex
__global__ void gat_exp(const int* __restrict__ src, const int* __restrict__ dst,
                        const float* __restrict__ as_, const float* __restrict__ ad_,
                        const unsigned* __restrict__ menc,
                        float* __restrict__ exbuf, float* __restrict__ denom,
                        int E, int total) {
    int id = blockIdx.x * blockDim.x + threadIdx.x;
    if (id >= total) return;
    int e = id >> 2, h = id & 3;
    int s, d;
    if (e < E) { s = src[e]; d = dst[e]; } else { s = d = e - E; }
    float v  = lrelu(as_[s * 4 + h] + ad_[d * 4 + h]);
    float ex = __expf(v - dec_f(menc[d * 4 + h]));
    exbuf[id] = ex;
    atomicAdd(&denom[d * 4 + h], ex);
}

// agg[d,f] += hW[s,f] * ex/denom
__global__ void gat_agg(const int* __restrict__ src, const int* __restrict__ dst,
                        const float* __restrict__ hW, const float* __restrict__ exbuf,
                        const float* __restrict__ denom, float* __restrict__ agg,
                        int E, int total) {
    int id = blockIdx.x * blockDim.x + threadIdx.x;
    if (id >= total) return;
    int e = id >> 7, f = id & 127, h = f >> 5;
    int s, d;
    if (e < E) { s = src[e]; d = dst[e]; } else { s = d = e - E; }
    float w = exbuf[(size_t)e * 4 + h] / denom[d * 4 + h];
    atomicAdd(&agg[(size_t)d * H128 + f], hW[(size_t)s * H128 + f] * w);
}

// ---------------------------------------------------------------------------
// pooling
// ---------------------------------------------------------------------------
__global__ void pool_scatter(const float* __restrict__ h, const int* __restrict__ batch,
                             float* __restrict__ pooled, float* __restrict__ cnt, int total) {
    int id = blockIdx.x * blockDim.x + threadIdx.x;
    if (id >= total) return;
    int n = id >> 7, f = id & 127;
    int g = batch[n];
    atomicAdd(&pooled[(size_t)g * 256 + 128 + f], h[id]);   // add-pool half
    if (f == 0) atomicAdd(&cnt[g], 1.0f);
}

__global__ void pool_mean(float* __restrict__ pooled, const float* __restrict__ cnt, int total) {
    int id = blockIdx.x * blockDim.x + threadIdx.x;
    if (id >= total) return;
    int g = id >> 7, f = id & 127;
    pooled[(size_t)g * 256 + f] = pooled[(size_t)g * 256 + 128 + f] / fmaxf(cnt[g], 1.0f);
}

// ---------------------------------------------------------------------------
// host orchestration
// ---------------------------------------------------------------------------
static inline int gblk(long long n, int b) { return (int)((n + b - 1) / b); }

extern "C" void kernel_launch(void* const* d_in, const int* in_sizes, int n_in,
                              void* d_out, int out_size, void* d_ws, size_t ws_size,
                              hipStream_t stream) {
    const float* x      = (const float*)d_in[0];
    const int*   eidx   = (const int*)  d_in[2];
    const int*   batch  = (const int*)  d_in[3];
    const float* node_W = (const float*)d_in[4];
    const float* node_b = (const float*)d_in[5];
    const float* gcn0_W = (const float*)d_in[10];
    const float* gcn0_b = (const float*)d_in[11];
    const float* gat1_W = (const float*)d_in[12];
    const float* gat1_as= (const float*)d_in[13];
    const float* gat1_ad= (const float*)d_in[14];
    const float* gat1_b = (const float*)d_in[15];
    const float* gcn2_W = (const float*)d_in[16];
    const float* gcn2_b = (const float*)d_in[17];
    const float* comb_W = (const float*)d_in[18];
    const float* comb_b = (const float*)d_in[19];
    const float* out_W1 = (const float*)d_in[20];
    const float* out_b1 = (const float*)d_in[21];
    const float* out_W2 = (const float*)d_in[22];
    const float* out_b2 = (const float*)d_in[23];
    const float* bn_g[3] = {(const float*)d_in[24], (const float*)d_in[28], (const float*)d_in[32]};
    const float* bn_b[3] = {(const float*)d_in[25], (const float*)d_in[29], (const float*)d_in[33]};
    const float* bn_m[3] = {(const float*)d_in[26], (const float*)d_in[30], (const float*)d_in[34]};
    const float* bn_v[3] = {(const float*)d_in[27], (const float*)d_in[31], (const float*)d_in[35]};

    const int E = in_sizes[2] / 2;
    const int N = in_sizes[3];
    const int G = out_size / H128;
    const int* src = eidx;
    const int* dst = eidx + E;

    // ---- workspace layout ----
    float* ws = (float*)d_ws;
    const size_t NB = (size_t)N * H128;
    float*    hA    = ws;               // N*128
    float*    hB    = hA + NB;          // N*128
    float*    hC    = hB + NB;          // N*128
    float*    dinv  = hC + NB;          // N
    float*    as_   = dinv + N;         // N*4
    float*    ad_   = as_ + (size_t)N*4;// N*4
    float*    denom = ad_ + (size_t)N*4;// N*4
    unsigned* menc  = (unsigned*)(denom + (size_t)N*4); // N*4
    float*    exbuf = (float*)menc + (size_t)N*4;       // (E+N)*4
    float*    pooled= exbuf + (size_t)(E + N)*4;        // G*256
    float*    cnt   = pooled + (size_t)G*256;           // G
    float*    g1    = cnt + G;                          // G*128
    float*    g2    = g1 + (size_t)G*H128;              // G*128
    // packed bf16 weights (fragment order)
    __bf16* wp_node = (__bf16*)(g2 + (size_t)G*H128);   // 64*128
    __bf16* wp_gcn0 = wp_node + (size_t)64 * H128;      // 128*128
    __bf16* wp_gat1 = wp_gcn0 + (size_t)H128 * H128;
    __bf16* wp_gcn2 = wp_gat1 + (size_t)H128 * H128;
    __bf16* wp_comb = wp_gcn2 + (size_t)H128 * H128;    // 256*128
    __bf16* wp_w1   = wp_comb + (size_t)256 * H128;
    __bf16* wp_w2   = wp_w1   + (size_t)H128 * H128;
    float*  outp    = (float*)d_out;

    const int TB = 256;
    const long long NE128  = (long long)E * H128;
    const long long NEN128 = (long long)(E + N) * H128;
    const long long N128   = (long long)N * H128;

    // ---- pack all weights to bf16 fragment layout (one-time, tiny) ----
    pack_w_bf16<<<gblk(64  * H128, TB), TB, 0, stream>>>(node_W, wp_node, 64  * H128);
    pack_w_bf16<<<gblk(H128* H128, TB), TB, 0, stream>>>(gcn0_W, wp_gcn0, H128* H128);
    pack_w_bf16<<<gblk(H128* H128, TB), TB, 0, stream>>>(gat1_W, wp_gat1, H128* H128);
    pack_w_bf16<<<gblk(H128* H128, TB), TB, 0, stream>>>(gcn2_W, wp_gcn2, H128* H128);
    pack_w_bf16<<<gblk(256 * H128, TB), TB, 0, stream>>>(comb_W, wp_comb, 256 * H128);
    pack_w_bf16<<<gblk(H128* H128, TB), TB, 0, stream>>>(out_W1, wp_w1,   H128* H128);
    pack_w_bf16<<<gblk(H128* H128, TB), TB, 0, stream>>>(out_W2, wp_w2,   H128* H128);

    // ---- input projection: hA = x @ node_W + node_b ----
    wmma_gemm128<1,0><<<N / 16, 256, 0, stream>>>(x, 64, wp_node, node_b, hA, 64);

    // ---- degrees (with self loops), dinv ----
    fill_f32<<<gblk(N, TB), TB, 0, stream>>>(dinv, 1.0f, N);
    deg_edge<<<gblk(E, TB), TB, 0, stream>>>(dst, dinv, E);
    deg_to_dinv<<<gblk(N, TB), TB, 0, stream>>>(dinv, N);

    // ---- layer 0: GCN (hA -> hC) ----
    wmma_gemm128<0,0><<<N / 16, 256, 0, stream>>>(hA, H128, wp_gcn0, nullptr, hB, H128);
    gcn_init<<<gblk(N128, TB), TB, 0, stream>>>(hB, dinv, hC, (int)N128);
    gcn_edge<<<gblk(NE128, TB), TB, 0, stream>>>(src, dst, hB, dinv, hC, (int)NE128);
    finalize_bn<<<gblk(N128, TB), TB, 0, stream>>>(hC, gcn0_b, bn_g[0], bn_b[0], bn_m[0], bn_v[0],
                                                   hA, hC, (int)N128);

    // ---- layer 1: GAT (hC -> hA) ----
    wmma_gemm128<0,0><<<N / 16, 256, 0, stream>>>(hC, H128, wp_gat1, nullptr, hB, H128);
    gat_alpha<<<gblk((long long)N * 4, TB), TB, 0, stream>>>(hB, gat1_as, gat1_ad, as_, ad_, N * 4);
    fill_f32<<<gblk((long long)N * 4, TB), TB, 0, stream>>>((float*)menc, 0.0f, N * 4); // enc 0 == -inf
    fill_f32<<<gblk((long long)N * 4, TB), TB, 0, stream>>>(denom, 0.0f, N * 4);
    gat_max<<<gblk((long long)(E + N) * 4, TB), TB, 0, stream>>>(src, dst, as_, ad_, menc, E, (E + N) * 4);
    gat_exp<<<gblk((long long)(E + N) * 4, TB), TB, 0, stream>>>(src, dst, as_, ad_, menc, exbuf, denom, E, (E + N) * 4);
    fill_f32<<<gblk(N128, TB), TB, 0, stream>>>(hA, 0.0f, (int)N128);
    gat_agg<<<gblk(NEN128, TB), TB, 0, stream>>>(src, dst, hB, exbuf, denom, hA, E, (int)NEN128);
    finalize_bn<<<gblk(N128, TB), TB, 0, stream>>>(hA, gat1_b, bn_g[1], bn_b[1], bn_m[1], bn_v[1],
                                                   hC, hA, (int)N128);

    // ---- layer 2: GCN (hA -> hC) ----
    wmma_gemm128<0,0><<<N / 16, 256, 0, stream>>>(hA, H128, wp_gcn2, nullptr, hB, H128);
    gcn_init<<<gblk(N128, TB), TB, 0, stream>>>(hB, dinv, hC, (int)N128);
    gcn_edge<<<gblk(NE128, TB), TB, 0, stream>>>(src, dst, hB, dinv, hC, (int)NE128);
    finalize_bn<<<gblk(N128, TB), TB, 0, stream>>>(hC, gcn2_b, bn_g[2], bn_b[2], bn_m[2], bn_v[2],
                                                   hA, hC, (int)N128);

    // ---- pooling (mean || add) -> [G,256] ----
    fill_f32<<<gblk((long long)G * 256, TB), TB, 0, stream>>>(pooled, 0.0f, G * 256);
    fill_f32<<<gblk(G, TB), TB, 0, stream>>>(cnt, 0.0f, G);
    pool_scatter<<<gblk(N128, TB), TB, 0, stream>>>(hC, batch, pooled, cnt, (int)N128);
    pool_mean<<<gblk((long long)G * H128, TB), TB, 0, stream>>>(pooled, cnt, G * H128);

    // ---- graph head ----
    wmma_gemm128<1,0><<<G / 16, 256, 0, stream>>>(pooled, 256, wp_comb, comb_b, g1, 256);
    wmma_gemm128<1,1><<<G / 16, 256, 0, stream>>>(g1, H128, wp_w1, out_b1, g2, H128);
    wmma_gemm128<1,0><<<G / 16, 256, 0, stream>>>(g2, H128, wp_w2, out_b2, outp, H128);
}
